// LSTMEncoderDecoderWrapper_40166534152243
// MI455X (gfx1250) — matrix operand order repaired and checked by reading, not compiled, not run on hardware
//
#include <hip/hip_runtime.h>
#include <hip/hip_bf16.h>

// ---------------- problem constants ----------------
#define B_    128
#define S_    256
#define I_    128
#define H_    1024
#define TOUT_ 128
#define FOUT_ 64
#define G4H   4096          // 4*H
#define KENC  1152          // I + H
#define KDEC  1088          // FOUT + H
#define KBE   36            // KENC/32
#define KBD   34            // KDEC/32
#define KBH   32            // H/32

typedef __bf16 v16bf __attribute__((ext_vector_type(16)));
typedef float  v8f   __attribute__((ext_vector_type(8)));

__device__ __forceinline__ unsigned short f2bf(float f) {
  unsigned x = __float_as_uint(f);
  unsigned r = x + 0x7FFFu + ((x >> 16) & 1u);   // round-to-nearest-even
  return (unsigned short)(r >> 16);
}

// gfx1250 async global->LDS copy, 16 bytes per lane, tracked by ASYNCcnt.
__device__ __forceinline__ void async_ld16(unsigned short* lds_dst,
                                           const unsigned short* g_src) {
  unsigned lds_addr = (unsigned)(unsigned long long)lds_dst; // addr[31:0] = LDS offset
  asm volatile("global_load_async_to_lds_b128 %0, %1, off"
               :: "v"(lds_addr), "v"(g_src)
               : "memory");
}
__device__ __forceinline__ void wait_asynccnt_4() {
  asm volatile("s_wait_asynccnt 0x4" ::: "memory");
}
__device__ __forceinline__ void wait_asynccnt_0() {
  asm volatile("s_wait_asynccnt 0x0" ::: "memory");
}

// ---------------------------------------------------------------------------
// Pack a weight matrix Wcat[n][k] (n in [0,N), k in [0,K)) into the WMMA
// B-fragment layout: idx = ((n_blk*KB + kb)*32 + lane)*16 + e
//   n = n_blk*16 + (lane&15); half = lane>>4
//   k = kb*32 + (e&7) + 16*(e>>3) + 8*half   (same k order as A fragment)
// Wcat rows: k < Kf -> W1[n*Kf+k], else W2[n*(K-Kf)+(k-Kf)]
// ---------------------------------------------------------------------------
__global__ __launch_bounds__(256) void pack_B_kernel(
    unsigned short* __restrict__ out, const float* __restrict__ W1,
    const float* __restrict__ W2, int Kf, int K, int KB, int total) {
  int tid = blockIdx.x * 256 + threadIdx.x;
  if (tid >= total) return;
  int e    = tid & 15;
  int lane = (tid >> 4) & 31;
  int blk  = tid >> 9;
  int kb   = blk % KB;
  int nblk = blk / KB;
  int n = nblk * 16 + (lane & 15);
  int k = kb * 32 + (e & 7) + ((e >> 3) << 4) + ((lane >> 4) << 3);
  float v = (k < Kf) ? W1[(long)n * Kf + k]
                     : W2[(long)n * (K - Kf) + (k - Kf)];
  out[tid] = f2bf(v);
}

// A-fragment layout (16-bit A 16x32, ISA 7.12.2):
//   idx = ((kb*8 + m_blk)*32 + lane)*16 + e
//   m = m_blk*16 + (lane&15); half = lane>>4
//   kk = (e&7) + 16*(e>>3) + 8*half ;  k = kb*32 + kk
__global__ __launch_bounds__(256) void pack_x_kernel(
    unsigned short* __restrict__ Apack, const float* __restrict__ x, int s) {
  int tid = blockIdx.x * 256 + threadIdx.x;          // 16384 total (kb 0..3)
  int e = tid & 15, lane = (tid >> 4) & 31;
  int mblk = (tid >> 9) & 7, kb = tid >> 12;
  int m  = mblk * 16 + (lane & 15);
  int kk = (e & 7) + ((e >> 3) << 4) + ((lane >> 4) << 3);
  int k  = kb * 32 + kk;
  Apack[tid] = f2bf(x[((long)m * S_ + s) * I_ + k]);
}

__global__ __launch_bounds__(256) void pack_y_kernel(
    unsigned short* __restrict__ Apack, const float* __restrict__ y,
    int rowStride) {
  int tid = blockIdx.x * 256 + threadIdx.x;          // 8192 total (kb 0..1)
  int e = tid & 15, lane = (tid >> 4) & 31;
  int mblk = (tid >> 9) & 7, kb = tid >> 12;
  int m  = mblk * 16 + (lane & 15);
  int kk = (e & 7) + ((e >> 3) << 4) + ((lane >> 4) << 3);
  int k  = kb * 32 + kk;
  Apack[tid] = f2bf(y[(long)m * rowStride + k]);
}

__global__ __launch_bounds__(256) void zero_init_kernel(
    unsigned short* __restrict__ a, int na, float* __restrict__ c, int nc) {
  int tid = blockIdx.x * 256 + threadIdx.x;
  if (tid < na) a[tid] = 0;
  if (tid < nc) c[tid] = 0.0f;
}

// ---------------------------------------------------------------------------
// gates[128 x 4096] = A[128 x K] * Wcat^T   (bf16 WMMA, f32 accum)
// grid.x = 32 (N tiles of 128), block = 256 (8 waves); wave = M slice of 16.
// Per k-step, the whole workgroup async-copies the A tile (8 KB) and the
// shared B tile (8 KB) into double-buffered LDS (ASYNCcnt pipeline).  All
// 9 fragments are preloaded to VGPRs (one s_wait_dscnt), then 8 WMMAs issue
// back-to-back into the matrix pipe.
// ---------------------------------------------------------------------------
__global__ __launch_bounds__(256) void gemm_gates_kernel(
    const unsigned short* __restrict__ Apack,
    const unsigned short* __restrict__ Bpack, int KB,
    float* __restrict__ out, int Ntot) {
  __shared__ __attribute__((aligned(64))) unsigned short sA[2][4096]; // 8KB x2
  __shared__ __attribute__((aligned(64))) unsigned short sB[2][4096]; // 8KB x2
  const int tid  = threadIdx.x;
  const int lane = tid & 31;
  const int wave = tid >> 5;   // m_blk, and B-chunk owner during copy
  const int n0   = blockIdx.x * 8;

  // Stage one k-block into LDS buffer `buf`: 4 async instructions per wave.
  auto stage = [&](int buf, int kb) {
    // A tile: 4096 contiguous ushorts at Apack + kb*4096
    const unsigned short* ga = Apack + (long)kb * 4096 + tid * 16;
    unsigned short* la = &sA[buf][tid * 16];
    async_ld16(la, ga);
    async_ld16(la + 8, ga + 8);
    // B tile: chunk `wave` (512 ushorts) at Bpack + ((n0+wave)*KB + kb)*512
    const unsigned short* gb =
        Bpack + ((long)(n0 + wave) * KB + kb) * 512 + lane * 16;
    unsigned short* lb = &sB[buf][wave * 512 + lane * 16];
    async_ld16(lb, gb);
    async_ld16(lb + 8, gb + 8);
  };

  v8f acc[8] = {};
  // Preload fragments to registers, then 8 back-to-back WMMAs.
  auto compute = [&](int buf) {
    const v16bf a =
        *reinterpret_cast<const v16bf*>(&sA[buf][wave * 512 + lane * 16]);
    v16bf bfrag[8];
#pragma unroll
    for (int ns = 0; ns < 8; ++ns)
      bfrag[ns] =
          *reinterpret_cast<const v16bf*>(&sB[buf][ns * 512 + lane * 16]);
#pragma unroll
    for (int ns = 0; ns < 8; ++ns)
      acc[ns] = __builtin_amdgcn_wmma_f32_16x16x32_bf16(
          false, a, false, bfrag[ns], (short)0, acc[ns], false, false);
  };

  stage(0, 0);
  int kb = 0;
  for (; kb < KB - 1; ++kb) {
    const int buf = kb & 1;
    stage(buf ^ 1, kb + 1);
    wait_asynccnt_4();     // stage kb's 4 async ops retired (in-order)
    __syncthreads();
    compute(buf);
    __syncthreads();
  }
  wait_asynccnt_0();
  __syncthreads();
  compute(kb & 1);

  // C/D layout: VGPR r -> M = r (lanes 0-15) / 8+r (lanes 16-31), N = lane%16
  const int mrow = wave * 16 + ((lane >> 4) << 3);
  const int ncol = lane & 15;
#pragma unroll
  for (int ns = 0; ns < 8; ++ns)
#pragma unroll
    for (int r = 0; r < 8; ++r)
      out[(long)(mrow + r) * Ntot + (n0 + ns) * 16 + ncol] = acc[ns][r];
}

// out[128 x 64] = A[128 x 1024] * W^T + bias ; single workgroup, 8 waves.
__global__ __launch_bounds__(256) void gemm_small_kernel(
    const unsigned short* __restrict__ Apack,
    const unsigned short* __restrict__ Bpack, int KB,
    const float* __restrict__ bias, float* __restrict__ out, int rowStride) {
  const int lane = threadIdx.x & 31;
  const int wave = threadIdx.x >> 5;

  v8f acc[4] = {};
  for (int kb = 0; kb < KB; ++kb) {
    const v16bf a = *reinterpret_cast<const v16bf*>(
        Apack + (((long)kb * 8 + wave) * 32 + lane) * 16);
    v16bf bfrag[4];
#pragma unroll
    for (int ns = 0; ns < 4; ++ns)
      bfrag[ns] = *reinterpret_cast<const v16bf*>(
          Bpack + (((long)ns * KB + kb) * 32 + lane) * 16);
#pragma unroll
    for (int ns = 0; ns < 4; ++ns)
      acc[ns] = __builtin_amdgcn_wmma_f32_16x16x32_bf16(
          false, a, false, bfrag[ns], (short)0, acc[ns], false, false);
  }
  const int mrow = wave * 16 + ((lane >> 4) << 3);
  const int ncol = lane & 15;
#pragma unroll
  for (int ns = 0; ns < 4; ++ns)
#pragma unroll
    for (int r = 0; r < 8; ++r)
      out[(long)(mrow + r) * rowStride + ns * 16 + ncol] =
          acc[ns][r] + bias[ns * 16 + ncol];
}

// ---------------------------------------------------------------------------
// LSTM pointwise update: gates (pre-bias) -> c update, h -> bf16 A-fragment
// pack at k = j + koff.  PyTorch gate order i,f,g,o.
// ---------------------------------------------------------------------------
__global__ __launch_bounds__(256) void lstm_update_kernel(
    const float* __restrict__ gates, const float* __restrict__ b_ih,
    const float* __restrict__ b_hh, float* __restrict__ c,
    unsigned short* __restrict__ Apack, int koff) {
  int tid = blockIdx.x * 256 + threadIdx.x;   // B*H = 131072
  int j = tid & (H_ - 1);
  int b = tid >> 10;
  const float* g = gates + (long)b * G4H;
  float gi = g[j]            + b_ih[j]            + b_hh[j];
  float gf = g[H_ + j]       + b_ih[H_ + j]       + b_hh[H_ + j];
  float gg = g[2 * H_ + j]   + b_ih[2 * H_ + j]   + b_hh[2 * H_ + j];
  float go = g[3 * H_ + j]   + b_ih[3 * H_ + j]   + b_hh[3 * H_ + j];
  float i_ = 1.0f / (1.0f + __expf(-gi));
  float f_ = 1.0f / (1.0f + __expf(-gf));
  float g_ = tanhf(gg);
  float o_ = 1.0f / (1.0f + __expf(-go));
  float cn = f_ * c[tid] + i_ * g_;
  c[tid] = cn;
  float hn = o_ * tanhf(cn);
  // A-fragment scatter
  int kr = j + koff;
  int kb = kr >> 5, kk = kr & 31;
  int half = (kk >> 3) & 1;
  int lane = (b & 15) + (half << 4);
  int e = (kk & 7) + ((kk >> 4) << 3);
  Apack[(((long)kb * 8 + (b >> 4)) * 32 + lane) * 16 + e] = f2bf(hn);
}

// ---------------------------------------------------------------------------
extern "C" void kernel_launch(void* const* d_in, const int* in_sizes, int n_in,
                              void* d_out, int out_size, void* d_ws, size_t ws_size,
                              hipStream_t stream) {
  (void)in_sizes; (void)n_in; (void)out_size; (void)ws_size;
  const float* x        = (const float*)d_in[0];
  const float* enc_W_ih = (const float*)d_in[1];
  const float* enc_W_hh = (const float*)d_in[2];
  const float* enc_b_ih = (const float*)d_in[3];
  const float* enc_b_hh = (const float*)d_in[4];
  const float* dense_W  = (const float*)d_in[5];
  const float* dense_b  = (const float*)d_in[6];
  const float* dec_W_ih = (const float*)d_in[7];
  const float* dec_W_hh = (const float*)d_in[8];
  const float* dec_b_ih = (const float*)d_in[9];
  const float* dec_b_hh = (const float*)d_in[10];
  const float* out_W    = (const float*)d_in[11];
  const float* out_b    = (const float*)d_in[12];
  float* outp = (float*)d_out;

  // ---- workspace carve-up (256B aligned) ----
  char* ws = (char*)d_ws;
  size_t off = 0;
  auto carve = [&](size_t bytes) -> void* {
    void* p = ws + off;
    off = (off + bytes + 255) & ~(size_t)255;
    return p;
  };
  const int encBtot = (G4H / 16) * KBE * 512;   // 4,718,592
  const int decBtot = (G4H / 16) * KBD * 512;   // 4,456,448
  const int smlBtot = (FOUT_ / 16) * KBH * 512; // 65,536
  unsigned short* WencP   = (unsigned short*)carve((size_t)encBtot * 2);
  unsigned short* WdecP   = (unsigned short*)carve((size_t)decBtot * 2);
  unsigned short* WoutP   = (unsigned short*)carve((size_t)smlBtot * 2);
  unsigned short* WdenseP = (unsigned short*)carve((size_t)smlBtot * 2);
  unsigned short* Aenc    = (unsigned short*)carve((size_t)B_ * KENC * 2);
  unsigned short* Adec    = (unsigned short*)carve((size_t)B_ * KDEC * 2);
  float* gates = (float*)carve((size_t)B_ * G4H * 4);
  float* cbuf  = (float*)carve((size_t)B_ * H_ * 4);
  float* ybuf  = (float*)carve((size_t)B_ * FOUT_ * 4);

  // ---- one-time packs ----
  pack_B_kernel<<<encBtot / 256, 256, 0, stream>>>(WencP, enc_W_ih, enc_W_hh,
                                                   I_, KENC, KBE, encBtot);
  pack_B_kernel<<<decBtot / 256, 256, 0, stream>>>(WdecP, dec_W_ih, dec_W_hh,
                                                   FOUT_, KDEC, KBD, decBtot);
  pack_B_kernel<<<smlBtot / 256, 256, 0, stream>>>(WoutP, out_W, out_W,
                                                   H_, H_, KBH, smlBtot);
  pack_B_kernel<<<smlBtot / 256, 256, 0, stream>>>(WdenseP, dense_W, dense_W,
                                                   H_, H_, KBH, smlBtot);
  // h0 = 0 inside Aenc (k-blocks 4..35) and c0 = 0
  zero_init_kernel<<<512, 256, 0, stream>>>(Aenc + 4 * 4096, 32 * 4096,
                                            cbuf, B_ * H_);

  // ---- encoder: 256 sequential steps ----
  for (int s = 0; s < S_; ++s) {
    pack_x_kernel<<<64, 256, 0, stream>>>(Aenc, x, s);
    gemm_gates_kernel<<<32, 256, 0, stream>>>(Aenc, WencP, KBE, gates, G4H);
    lstm_update_kernel<<<512, 256, 0, stream>>>(gates, enc_b_ih, enc_b_hh,
                                                cbuf, Aenc, I_);
  }

  // ---- dense bridge: y0 = h_n @ dense_W^T + dense_b ----
  gemm_small_kernel<<<1, 256, 0, stream>>>(Aenc + 4 * 4096, WdenseP, KBH,
                                           dense_b, ybuf, FOUT_);
  // move packed h_n into decoder A buffer (k-blocks 2..33)
  hipMemcpyAsync(Adec + 2 * 4096, Aenc + 4 * 4096, (size_t)32 * 4096 * 2,
                 hipMemcpyDeviceToDevice, stream);
  pack_y_kernel<<<32, 256, 0, stream>>>(Adec, ybuf, FOUT_);

  // ---- decoder: 128 autoregressive steps ----
  for (int t = 0; t < TOUT_; ++t) {
    gemm_gates_kernel<<<32, 256, 0, stream>>>(Adec, WdecP, KBD, gates, G4H);
    lstm_update_kernel<<<512, 256, 0, stream>>>(gates, dec_b_ih, dec_b_hh,
                                                cbuf, Adec, FOUT_);
    gemm_small_kernel<<<1, 256, 0, stream>>>(Adec + 2 * 4096, WoutP, KBH,
                                             out_b, outp + (size_t)t * FOUT_,
                                             TOUT_ * FOUT_);
    pack_y_kernel<<<32, 256, 0, stream>>>(Adec, outp + (size_t)t * FOUT_,
                                          TOUT_ * FOUT_);
  }
}
